// ALSTM_90778428769074
// MI455X (gfx1250) — matrix operand order repaired and checked by reading, not compiled, not run on hardware
//
#include <hip/hip_runtime.h>

typedef __attribute__((ext_vector_type(16))) _Float16 v16h;
typedef __attribute__((ext_vector_type(8)))  _Float16 v8h;
typedef __attribute__((ext_vector_type(8)))  float    v8f;

#define B_   32
#define T_   256
#define M_   16
#define D_   128
#define H_   128
#define G4   512          // 4*H (gate order: i,f,c,o)
#define ROWS (B_ * T_)    // 8192

// ---------------------------------------------------------------------------
// Kernel 1: w_x[d] = sum_h W_a[d,h] for d < D  (h-part + bias cancel in softmax)
// ---------------------------------------------------------------------------
__global__ void k_wx(const float* __restrict__ Wa, float* __restrict__ wx) {
    int d = threadIdx.x;                 // launched with 128 threads
    const float* row = Wa + d * H_;
    float s = 0.f;
    for (int h = 0; h < H_; ++h) s += row[h];
    wx[d] = s;
}

// ---------------------------------------------------------------------------
// Kernel 2: pack weights transposed to f16:  W4T[n][d] = W_g[d][n%128],
//           UT[n][d] = U_g[d][n%128], b4[n] = b_g[n%128],  n = g*128+col
// Transposed layout gives contiguous per-lane B fragments for WMMA.
// ---------------------------------------------------------------------------
__global__ void k_pack(const float* Wi, const float* Ui, const float* bi,
                       const float* Wf, const float* Uf, const float* bf,
                       const float* Wc, const float* Uc, const float* bc,
                       const float* Wo, const float* Uo, const float* bo,
                       _Float16* __restrict__ W4T, _Float16* __restrict__ UT,
                       float* __restrict__ b4) {
    int idx = blockIdx.x * blockDim.x + threadIdx.x;   // 0 .. 512*128-1
    int n = idx >> 7;
    int d = idx & 127;
    int g = n >> 7;
    int col = n & 127;
    const float* W = (g == 0) ? Wi : (g == 1) ? Wf : (g == 2) ? Wc : Wo;
    const float* U = (g == 0) ? Ui : (g == 1) ? Uf : (g == 2) ? Uc : Uo;
    W4T[n * D_ + d] = (_Float16)W[d * H_ + col];
    UT [n * D_ + d] = (_Float16)U[d * H_ + col];
    if (d == 0) {
        const float* bb = (g == 0) ? bi : (g == 1) ? bf : (g == 2) ? bc : bo;
        b4[n] = bb[col];
    }
}

// ---------------------------------------------------------------------------
// Kernel 3: attention pooling, one block per (b,t). Bandwidth bound (64 MiB).
// scores[m] = dot(x[b,t,m,:], w_x); softmax over m; new_x = sum_m sm*x (f16 out)
// ---------------------------------------------------------------------------
__global__ void k_attn(const float* __restrict__ x, const float* __restrict__ wx,
                       _Float16* __restrict__ newx) {
    __shared__ float xs[M_][D_];
    __shared__ float wxs[D_];
    __shared__ float sm[M_];
    int tid = threadIdx.x;                               // 256 threads
    const float* xt = x + (size_t)blockIdx.x * (M_ * D_);
    for (int i = tid; i < M_ * D_; i += 256) xs[i >> 7][i & 127] = xt[i];
    if (tid < D_) wxs[tid] = wx[tid];
    __syncthreads();

    int m = tid >> 4, l = tid & 15;                      // 16 lanes per row
    float p = 0.f;
    for (int i = 0; i < 8; ++i) p += xs[m][l * 8 + i] * wxs[l * 8 + i];
    for (int off = 8; off; off >>= 1) p += __shfl_xor(p, off, 16);
    if (l == 0) sm[m] = p;
    __syncthreads();

    if (tid == 0) {                                      // 16-elem softmax
        float mx = sm[0];
        for (int i = 1; i < M_; ++i) mx = fmaxf(mx, sm[i]);
        float s = 0.f;
        for (int i = 0; i < M_; ++i) { float e = expf(sm[i] - mx); sm[i] = e; s += e; }
        float inv = 1.f / s;
        for (int i = 0; i < M_; ++i) sm[i] *= inv;
    }
    __syncthreads();

    if (tid < D_) {
        float acc = 0.f;
        for (int i = 0; i < M_; ++i) acc += sm[i] * xs[i][tid];
        newx[(size_t)blockIdx.x * D_ + tid] = (_Float16)acc;
    }
}

// ---------------------------------------------------------------------------
// Kernel 4: Zx = new_x(8192x128,f16) @ W4T^T(128x512,f16) + b4, f32 out.
// 8 waves/block, 1 16x16 tile per wave, K=128 in 4 WMMA steps.
// ---------------------------------------------------------------------------
__global__ void k_gemm_x(const _Float16* __restrict__ A,
                         const _Float16* __restrict__ Bt,   // (512 x 128) transposed
                         const float* __restrict__ b4,
                         float* __restrict__ Z) {
    int wave = threadIdx.x >> 5;
    int lane = threadIdx.x & 31;
    int l16 = lane & 15;
    int hi = lane >> 4;
    int rt = blockIdx.x;                  // row tile 0..511
    int ct = blockIdx.y * 8 + wave;       // col tile 0..31
    int row = rt * 16 + l16;
    int col = ct * 16 + l16;
    v8f acc = {};
    for (int k0 = 0; k0 < D_; k0 += 32) {
        // A fragment: rows l16, contiguous K16 at k0+hi*16 (matches B)
        const _Float16* ap = A + (size_t)row * D_ + k0 + hi * 16;
        v8h a0 = *(const v8h*)(ap);
        v8h a1 = *(const v8h*)(ap + 8);
        // B fragment: col l16, contiguous K16 at k0+hi*16 (same k order as A)
        const _Float16* bp = Bt + (size_t)col * D_ + k0 + hi * 16;
        v8h b0 = *(const v8h*)(bp);
        v8h b1 = *(const v8h*)(bp + 8);
        v16h av, bv;
        for (int i = 0; i < 8; ++i) { av[i] = a0[i]; av[8 + i] = a1[i];
                                      bv[i] = b0[i]; bv[8 + i] = b1[i]; }
        acc = __builtin_amdgcn_wmma_f32_16x16x32_f16(false, av, false, bv,
                                                     (short)0, acc, false, false);
    }
    float bias = b4[col];
    int rbase = rt * 16 + hi * 8;         // C/D layout: lane>=16 -> M += 8
    for (int r = 0; r < 8; ++r)
        Z[(size_t)(rbase + r) * G4 + col] = acc[r] + bias;
}

// ---------------------------------------------------------------------------
// Kernel 5: sequential LSTM recurrence. One persistent workgroup, 32 waves.
// U^T resident in LDS (128 KB); per step: gates = Zx[t] + h @ U via WMMA,
// then elementwise hard-sigmoid/tanh update of c,h.
// LDS: UT 128K | hC(f16) 8K | c(f32) 16K | gates(f32) 64K = 216 KB (<320 KB/WGP)
// ---------------------------------------------------------------------------
__global__ __launch_bounds__(1024) void k_recur(const _Float16* __restrict__ UTg,
                                                const float* __restrict__ Z,
                                                float* __restrict__ out) {
    extern __shared__ char smem[];
    _Float16* UT = (_Float16*)smem;                              // 512*128 f16
    _Float16* hC = (_Float16*)(smem + 131072);                   // 32*128 f16
    float*    cS = (float*)(smem + 131072 + 8192);               // 32*128 f32
    float*    gS = (float*)(smem + 131072 + 8192 + 16384);       // 32*512 f32
    int tid = threadIdx.x;

    { // stage U^T into LDS, zero h and c
        const unsigned int* src = (const unsigned int*)UTg;
        unsigned int* dst = (unsigned int*)UT;
        for (int i = tid; i < 512 * 128 / 2; i += 1024) dst[i] = src[i];
        unsigned int* hU = (unsigned int*)hC;
        for (int i = tid; i < 32 * 128 / 2; i += 1024) hU[i] = 0u;
        for (int i = tid; i < 32 * 128; i += 1024) cS[i] = 0.f;
    }
    __syncthreads();

    int wave = tid >> 5;
    int lane = tid & 31;
    int l16 = lane & 15;
    int hi = lane >> 4;

    for (int t = 0; t < T_; ++t) {
        for (int j = 0; j < 2; ++j) {                 // 2 tiles per wave
            int tile = wave * 2 + j;                  // 64 tiles: 2 row x 32 col
            int mt = tile >> 5;
            int ct = tile & 31;
            int arow = mt * 16 + l16;
            int col = ct * 16 + l16;
            v8f acc = {};
            for (int k0 = 0; k0 < H_; k0 += 32) {
                const _Float16* ap = hC + arow * H_ + k0 + hi * 16;
                v8h a0 = *(const v8h*)ap;
                v8h a1 = *(const v8h*)(ap + 8);
                const _Float16* bp = UT + col * H_ + k0 + hi * 16;
                v8h b0 = *(const v8h*)bp;
                v8h b1 = *(const v8h*)(bp + 8);
                v16h av, bv;
                for (int i = 0; i < 8; ++i) { av[i] = a0[i]; av[8 + i] = a1[i];
                                              bv[i] = b0[i]; bv[8 + i] = b1[i]; }
                acc = __builtin_amdgcn_wmma_f32_16x16x32_f16(false, av, false, bv,
                                                             (short)0, acc, false, false);
            }
            int b0r = mt * 16 + hi * 8;
            for (int r = 0; r < 8; ++r) {
                int b = b0r + r;
                gS[b * G4 + col] = acc[r] + Z[((size_t)b * T_ + t) * G4 + col];
            }
        }
        __syncthreads();

        for (int e = tid; e < B_ * H_; e += 1024) {   // 4 elements per thread
            int b = e >> 7, jj = e & 127;
            float ig = gS[b * G4 + jj];
            float fg = gS[b * G4 + 128 + jj];
            float cg = gS[b * G4 + 256 + jj];
            float og = gS[b * G4 + 384 + jj];
            ig = fminf(fmaxf(0.2f * ig + 0.5f, 0.f), 1.f);   // hard_sigmoid
            fg = fminf(fmaxf(0.2f * fg + 0.5f, 0.f), 1.f);
            og = fminf(fmaxf(0.2f * og + 0.5f, 0.f), 1.f);
            float cn = fg * cS[e] + ig * tanhf(cg);
            float hn = og * tanhf(cn);
            cS[e] = cn;
            hC[e] = (_Float16)hn;
            if (t == T_ - 1) out[e] = hn;
        }
        __syncthreads();
    }
}

// ---------------------------------------------------------------------------
extern "C" void kernel_launch(void* const* d_in, const int* in_sizes, int n_in,
                              void* d_out, int out_size, void* d_ws, size_t ws_size,
                              hipStream_t stream) {
    const float* x  = (const float*)d_in[0];
    const float* Wa = (const float*)d_in[1];
    // d_in[2] = b_a : cancels in softmax, unused
    const float* Wi = (const float*)d_in[3],  *Ui = (const float*)d_in[4],
               * bi = (const float*)d_in[5];
    const float* Wf = (const float*)d_in[6],  *Uf = (const float*)d_in[7],
               * bf = (const float*)d_in[8];
    const float* Wc = (const float*)d_in[9],  *Uc = (const float*)d_in[10],
               * bc = (const float*)d_in[11];
    const float* Wo = (const float*)d_in[12], *Uo = (const float*)d_in[13],
               * bo = (const float*)d_in[14];

    // Workspace carve (all offsets 256B aligned); total ~18.3 MB
    char* ws = (char*)d_ws;
    _Float16* newx = (_Float16*)ws;                              // 8192*128 f16 = 2 MiB
    _Float16* W4T  = (_Float16*)(ws + 2097152);                  // 512*128 f16 = 128 KiB
    _Float16* UT   = (_Float16*)(ws + 2097152 + 131072);         // 512*128 f16 = 128 KiB
    float*    b4   = (float*)   (ws + 2097152 + 262144);         // 512 f32 (padded 2 KiB)
    float*    wx   = (float*)   (ws + 2097152 + 262144 + 2048);  // 128 f32 (padded 512 B)
    float*    Z    = (float*)   (ws + 2097152 + 262144 + 2560);  // 8192*512 f32 = 16 MiB

    k_wx  <<<1, 128, 0, stream>>>(Wa, wx);
    k_pack<<<256, 256, 0, stream>>>(Wi, Ui, bi, Wf, Uf, bf, Wc, Uc, bc,
                                    Wo, Uo, bo, W4T, UT, b4);
    k_attn<<<ROWS, 256, 0, stream>>>(x, wx, newx);
    dim3 g2(ROWS / 16, 4);
    k_gemm_x<<<g2, 256, 0, stream>>>(newx, W4T, b4, Z);
    k_recur<<<1, 1024, 131072 + 8192 + 16384 + 65536, stream>>>(UT, Z, (float*)d_out);
}
